// SparseAttention_24558622998727
// MI455X (gfx1250) — matrix operand order repaired and checked by reading, not compiled, not run on hardware
//
#include <hip/hip_runtime.h>

// ---------------------------------------------------------------------------
// Sparse top-k attention for MI455X (gfx1250, wave32, WMMA bf16).
// B=8, Q=K=2048, D=256, TOP_K=64.
// Phase 1 computes S^T = K_chunk x Q^T; the A (K row) fragment is
// per-lane contiguous in LDS (ds_load_b128); B is gathered per element from Qt[d][q].
// Phases 2/3 sweep the score buffer at uint4 (8 x bf16) granularity.
// ---------------------------------------------------------------------------

typedef __attribute__((ext_vector_type(16))) __bf16 v16bf;
typedef __attribute__((ext_vector_type(8)))  float  v8f;

#define B_      8
#define Qn      2048
#define Kn      2048
#define Dn      256
#define TOPK    64
#define NEGV    (-1000000.0f)

#define QTILE   16
#define KCHUNK  64
#define THREADS 128          // 4 waves (wave32)
#define DPAD    (Dn + 8)     // 264 u16 = 528 B rows: 16B-aligned, banks 4*row
#define KPAD    (Kn + 8)     // 2056 u16 = 4112 B rows: 16B-aligned
#define QTPAD   (QTILE + 8)  // 24 u16 = 48 B rows: 16B-aligned

__device__ __forceinline__ unsigned short f2bf(float f) {
  unsigned int x = __float_as_uint(f);
  x += 0x7FFFu + ((x >> 16) & 1u);          // round-to-nearest-even
  return (unsigned short)(x >> 16);
}
__device__ __forceinline__ float bf2f(unsigned short h) {
  return __uint_as_float(((unsigned int)h) << 16);
}
// order-preserving uint16 image of bf16 (monotone increasing)
__device__ __forceinline__ unsigned int ordkey(unsigned short bits) {
  return (bits & 0x8000u) ? (unsigned int)((~bits) & 0xFFFFu)
                          : (unsigned int)(bits | 0x8000u);
}
__device__ __forceinline__ float keyval(unsigned int u) {
  unsigned short bits = (u & 0x8000u) ? (unsigned short)(u ^ 0x8000u)
                                      : (unsigned short)((~u) & 0xFFFFu);
  return bf2f(bits);
}
// Build a 16-element bf16 fragment from two contiguous 8-element LDS runs.
__device__ __forceinline__ v16bf load_frag2(const unsigned short* p0,
                                            const unsigned short* p1) {
  union { uint4 u[2]; v16bf v; } f;
  f.u[0] = *(const uint4*)p0;
  f.u[1] = *(const uint4*)p1;
  return f.v;
}

__global__ __launch_bounds__(THREADS)
void sparse_topk_attn_gfx1250(const float* __restrict__ q,
                              const float* __restrict__ kmat,
                              const float* __restrict__ vmat,
                              const int*   __restrict__ vlen,
                              float*       __restrict__ out) {
  __shared__ __align__(16) unsigned short Qt[Dn][QTPAD];     // 12 KB bf16 Q^T
  __shared__ __align__(16) unsigned short KV[KCHUNK][DPAD];  // 33 KB bf16 K/V chunk
  __shared__ __align__(16) unsigned short Sb[QTILE][KPAD];   // 64.25 KB bf16 scores/probs
  __shared__ unsigned int hist[4][256];                      // radix histograms (1/wave)
  __shared__ float rowKth[QTILE];
  __shared__ float rowMax[QTILE];
  __shared__ float rowSum[QTILE];

  const int tid   = threadIdx.x;
  const int lane  = tid & 31;
  const int wave  = tid >> 5;
  const int b     = blockIdx.y;
  const int qbase = blockIdx.x * QTILE;
  const int vl    = vlen[b];
  const float scale = 0.0625f;                   // 1/sqrt(256)

  // ---- stage Q^T (d-major: Qt[d][q]) once, coalesced dword gathers ---------
  {
    const float* qg = q + ((size_t)b * Qn + qbase) * Dn;
#pragma unroll
    for (int it = 0; it < (Dn * (QTILE / 8)) / THREADS; ++it) {   // 4 iters
      int task = tid + it * THREADS;
      int d  = task & (Dn - 1);
      int q0 = (task >> 8) * 8;                  // 0 or 8
      union { unsigned short us[8]; uint4 u; } pk;
#pragma unroll
      for (int jj = 0; jj < 8; ++jj)
        pk.us[jj] = f2bf(qg[(size_t)(q0 + jj) * Dn + d]);
      *(uint4*)&Qt[d][q0] = pk.u;
    }
  }
  __syncthreads();

  // ---- phase 1: S^T tiles = (K_chunk x Q^T) * scale, masked bf16 into Sb ---
  for (int kc = 0; kc < Kn; kc += KCHUNK) {
    const float4* kg = (const float4*)(kmat + ((size_t)b * Kn + kc) * Dn);
#pragma unroll
    for (int it = 0; it < (KCHUNK * Dn / 4) / THREADS; ++it) {
      int i = tid + it * THREADS;
      int row = i >> 6, c4 = i & 63;
      float4 val = kg[row * 64 + c4];
      union { unsigned short us[4]; uint2 u2; } pk;
      pk.us[0] = f2bf(val.x); pk.us[1] = f2bf(val.y);
      pk.us[2] = f2bf(val.z); pk.us[3] = f2bf(val.w);
      *(uint2*)&KV[row][c4 * 4] = pk.u2;
    }
    if (kc + KCHUNK < Kn) {   // gfx1250 global_prefetch of next chunk
      const char* nxt = (const char*)(kmat + ((size_t)b * Kn + kc + KCHUNK) * Dn);
      __builtin_prefetch(nxt + (size_t)tid * 512, 0, 1);
    }
    __syncthreads();

    v8f acc = {};
    const int abase = (lane >= 16) ? 8 : 0;      // K-dim sub-offset per ISA A layout
    const int arow  = wave * 16 + (lane & 15);   // key row within chunk
#pragma unroll
    for (int kk = 0; kk < Dn; kk += 32) {
      v16bf af  = load_frag2(&KV[arow][kk + abase], &KV[arow][kk + abase + 16]);
      v16bf bfr;
#pragma unroll
      for (int s = 0; s < 16; ++s)
        bfr[s] = *(const __bf16*)&Qt[kk + abase + (s & 7) + ((s >> 3) << 4)][lane & 15];
      acc = __builtin_amdgcn_wmma_f32_16x16x32_bf16(
          false, af, false, bfr, (short)0, acc, false, false);
    }
    // D tile is (key x query): M = key-in-tile, N = query row
    const int qrow = lane & 15;
#pragma unroll
    for (int j = 0; j < 8; ++j) {
      const int key = kc + wave * 16 + abase + j;   // abase == mbase for C/D layout
      float s = acc[j] * scale;
      if (key >= vl) s = NEGV;
      Sb[qrow][key] = f2bf(s);
    }
    __syncthreads();
  }

  // ---- phase 2: per-row 64th-largest via 2-level radix select --------------
  // Sweeps run at uint4 (8 x bf16) granularity: rows are 16B-aligned.
  for (int rr = 0; rr < 4; ++rr) {
    const int r = wave + rr * 4;
    for (int i = lane; i < 256; i += 32) hist[wave][i] = 0;
    __syncthreads();
    unsigned int umax = 0;
    for (int g = lane; g < Kn / 8; g += 32) {
      union { uint4 u; unsigned short us[8]; } w;
      w.u = *(const uint4*)&Sb[r][g * 8];
#pragma unroll
      for (int e = 0; e < 8; ++e) {
        unsigned int u = ordkey(w.us[e]);
        umax = umax > u ? umax : u;
        atomicAdd(&hist[wave][u >> 8], 1u);
      }
    }
#pragma unroll
    for (int off = 16; off > 0; off >>= 1) {
      unsigned int o = __shfl_xor(umax, off, 32);
      umax = umax > o ? umax : o;
    }
    __syncthreads();
    unsigned int hb = 0, cumBefore = 0;
    if (lane == 0) {
      unsigned int cum = 0;
      for (int i2 = 255; i2 >= 0; --i2) {
        unsigned int c = hist[wave][i2];
        if (cum + c >= TOPK) { hb = (unsigned int)i2; cumBefore = cum; break; }
        cum += c;
      }
    }
    hb        = __shfl(hb, 0, 32);
    cumBefore = __shfl(cumBefore, 0, 32);
    __syncthreads();
    for (int i = lane; i < 256; i += 32) hist[wave][i] = 0;
    __syncthreads();
    for (int g = lane; g < Kn / 8; g += 32) {
      union { uint4 u; unsigned short us[8]; } w;
      w.u = *(const uint4*)&Sb[r][g * 8];
#pragma unroll
      for (int e = 0; e < 8; ++e) {
        unsigned int u = ordkey(w.us[e]);
        if ((u >> 8) == hb) atomicAdd(&hist[wave][u & 255], 1u);
      }
    }
    __syncthreads();
    if (lane == 0) {
      unsigned int cum = cumBefore, lb = 0;
      for (int i2 = 255; i2 >= 0; --i2) {
        cum += hist[wave][i2];
        if (cum >= TOPK) { lb = (unsigned int)i2; break; }
      }
      rowKth[r] = keyval((hb << 8) | lb);
      rowMax[r] = keyval(umax);
    }
    __syncthreads();
  }

  // ---- phase 3: thresholded exp in place (packed), row sums ----------------
  for (int rr = 0; rr < 4; ++rr) {
    const int r = wave + rr * 4;
    const float kth = rowKth[r];
    const float m   = rowMax[r];
    float sum = 0.f;
    for (int g = lane; g < Kn / 8; g += 32) {
      union { uint4 u; unsigned short us[8]; } w;
      w.u = *(const uint4*)&Sb[r][g * 8];
#pragma unroll
      for (int e = 0; e < 8; ++e) {
        float s  = bf2f(w.us[e]);
        float ev = (s >= kth) ? __expf(s - m) : 0.f;
        sum += ev;
        w.us[e] = f2bf(ev);
      }
      *(uint4*)&Sb[r][g * 8] = w.u;
    }
#pragma unroll
    for (int off = 16; off > 0; off >>= 1) sum += __shfl_xor(sum, off, 32);
    if (lane == 0) rowSum[r] = sum;
  }
  __syncthreads();

  // ---- phase 4: O = P V, normalize, store f32 ------------------------------
  v8f oacc[4] = {{}, {}, {}, {}};
  for (int kc = 0; kc < Kn; kc += KCHUNK) {
    const float4* vg = (const float4*)(vmat + ((size_t)b * Kn + kc) * Dn);
#pragma unroll
    for (int it = 0; it < (KCHUNK * Dn / 4) / THREADS; ++it) {
      int i = tid + it * THREADS;
      int row = i >> 6, c4 = i & 63;
      float4 val = vg[row * 64 + c4];
      union { unsigned short us[4]; uint2 u2; } pk;
      pk.us[0] = f2bf(val.x); pk.us[1] = f2bf(val.y);
      pk.us[2] = f2bf(val.z); pk.us[3] = f2bf(val.w);
      *(uint2*)&KV[row][c4 * 4] = pk.u2;
    }
    if (kc + KCHUNK < Kn) {
      const char* nxt = (const char*)(vmat + ((size_t)b * Kn + kc + KCHUNK) * Dn);
      __builtin_prefetch(nxt + (size_t)tid * 512, 0, 1);
    }
    __syncthreads();

    const int arow  = lane & 15;
    const int abase = (lane >= 16) ? 8 : 0;
#pragma unroll
    for (int kk = 0; kk < KCHUNK; kk += 32) {
      v16bf af = load_frag2(&Sb[arow][kc + kk + abase],
                            &Sb[arow][kc + kk + abase + 16]);
#pragma unroll
      for (int t = 0; t < 4; ++t) {
        const int col0 = wave * 64 + t * 16;
        v16bf bfr;
#pragma unroll
        for (int s = 0; s < 16; ++s)
          bfr[s] = *(const __bf16*)&KV[kk + abase + (s & 7) + ((s >> 3) << 4)][col0 + (lane & 15)];
        oacc[t] = __builtin_amdgcn_wmma_f32_16x16x32_bf16(
            false, af, false, bfr, (short)0, oacc[t], false, false);
      }
    }
    __syncthreads();
  }

  const int mbase = (lane >= 16) ? 8 : 0;
#pragma unroll
  for (int j = 0; j < 8; ++j) {
    const float inv = 1.0f / rowSum[mbase + j];
    const size_t rowoff = (((size_t)b * Qn) + qbase + mbase + j) * Dn;
#pragma unroll
    for (int t = 0; t < 4; ++t) {
      int n = wave * 64 + t * 16 + (lane & 15);
      out[rowoff + n] = oacc[t][j] * inv;
    }
  }
}

extern "C" void kernel_launch(void* const* d_in, const int* in_sizes, int n_in,
                              void* d_out, int out_size, void* d_ws, size_t ws_size,
                              hipStream_t stream) {
  (void)in_sizes; (void)n_in; (void)out_size; (void)d_ws; (void)ws_size;
  const float* q    = (const float*)d_in[0];
  const float* kmat = (const float*)d_in[1];
  const float* vmat = (const float*)d_in[2];
  const int*   vl   = (const int*)d_in[3];
  float* out = (float*)d_out;

  dim3 grid(Qn / QTILE, B_);   // 128 query tiles x 8 batches
  dim3 block(THREADS);
  hipLaunchKernelGGL(sparse_topk_attn_gfx1250, grid, block, 0, stream,
                     q, kmat, vmat, vl, out);
}
